// Model_40570261078612
// MI455X (gfx1250) — matrix operand order repaired and checked by reading, not compiled, not run on hardware
//
#include <hip/hip_runtime.h>
#include <hip/hip_bf16.h>
#include <cstddef>

// ---------------------------------------------------------------------------
// 16-layer linear-attention + soft-MoE LM.  All big matmuls via
// v_wmma_f32_16x16x32_bf16: 128x128 block tile, 32x64 wave tile (8 WMMA/wave
// per K-step), bf16 operands double-buffered in LDS (ping-pong, 1 barrier per
// K-step, global loads for tile i+1 overlap WMMA on tile i), fp32 accumulate,
// compile-time fused epilogues, strength-reduced staging pointers.
// B=2,T=1024,U=768,L=16,E=8,V=32000.
// ---------------------------------------------------------------------------

#define EPSF 1e-6f

typedef __attribute__((ext_vector_type(16))) __bf16 v16bf;
typedef __attribute__((ext_vector_type(8)))  float  v8f;
typedef __attribute__((ext_vector_type(4)))  __bf16 bf16x4;
typedef __attribute__((ext_vector_type(8)))  __bf16 bf16x8;

#define MBLK 128     // block tile M (4 waves x 32)
#define NBLK 128     // block tile N (2 waves x 64)
#define KB   32      // K step per WMMA
#define LDS_PAD 40   // padded LDS row (elements): 80B, 16B-aligned

// AMODE: 0 = A row-major, 1 = A transposed (A[k*lda+m]), 2 = gathered rows
// MODE (epilogue):
// 0: C = acc + bias
// 1: C = phi(acc+bias) = (z>0 ? z+1 : exp(z))            (elu+1)
// 2: C = sigmoid(acc+bias)
// 3: C = sigmoid(acc+bias) + res
// 4: C = acc/(aux[m]+EPS) + res                          (linear-attn combine)
// 5: C = aux[m*astride]*(acc+bias)                       (MoE init, expert 0)
// 6: C = C + aux[m*astride]*(acc+bias)                   (MoE accumulate)
// 7: C = acc * 2.0f                                      (logits / TEMP)
template <int AMODE, int MODE>
__launch_bounds__(256)
__global__ void gemm_wmma(const float* __restrict__ A, int lda,
                          const int* __restrict__ rowidx,
                          const float* __restrict__ B, int ldb,
                          const float* __restrict__ bias,
                          const float* __restrict__ aux, int astride,
                          const float* __restrict__ res,
                          float* __restrict__ C, int ldc,
                          int M, int N, int K) {
  __shared__ __bf16 As[2][MBLK][LDS_PAD];   // [buf][m][k]
  __shared__ __bf16 Bs[2][NBLK][LDS_PAD];   // [buf][n][k] (transposed)

  const int tid   = threadIdx.x;
  const int wave  = tid >> 5;
  const int lane  = tid & 31;
  const int waveM = wave >> 1;           // 0..3
  const int waveN = wave & 1;            // 0..1
  const int m0    = blockIdx.y * MBLK;
  const int n0    = blockIdx.x * NBLK;

  // ---- per-thread staging state: running pointers, advanced by KB each
  // stage call (strength-reduced; no per-step 64-bit multiplies). ----
  const float* aP[4];                    // AMODE 0/2: one float4-chunk each
  const float* aBase = nullptr;          // AMODE 1 running base
  int aoff[16];                          // AMODE 1 K-invariant offsets
  if (AMODE == 1) {
    aBase = A + m0;
#pragma unroll
    for (int i = 0; i < 16; ++i) {
      int c = i * 256 + tid;
      aoff[i] = (c >> 7) * lda + (c & 127);
    }
  } else {
#pragma unroll
    for (int i = 0; i < 4; ++i) {
      int c   = i * 256 + tid;
      int row = c >> 3;
      int kc  = (c & 7) * 4;
      int r   = (AMODE == 2) ? rowidx[m0 + row] : (m0 + row);
      aP[i] = A + (size_t)r * lda + kc;
    }
  }
  const size_t aStep = (size_t)KB * lda;
  const size_t bStep = (size_t)KB * ldb;
  // B: thread owns n-column nn = tid&127 in k-half kh = (tid>>7)*16.
  const float* bP = B + (size_t)((tid >> 7) * 16) * ldb + n0 + (tid & 127);

  // Stage one 128x32 A tile + 32x128 B tile (fp32 -> bf16) into buffer p,
  // then advance the running pointers.
  auto stage = [&](int p) {
    if (AMODE == 1) {
#pragma unroll
      for (int i = 0; i < 16; ++i) {
        int c   = i * 256 + tid;
        int row = c & 127;
        int kk  = c >> 7;
        As[p][row][kk] = (__bf16)aBase[aoff[i]];
      }
      aBase += aStep;
    } else {
#pragma unroll
      for (int i = 0; i < 4; ++i) {
        int c   = i * 256 + tid;
        int row = c >> 3;
        int kc  = (c & 7) * 4;
        const float4 f = *(const float4*)aP[i];
        aP[i] += KB;
        bf16x4 pk = {(__bf16)f.x, (__bf16)f.y, (__bf16)f.z, (__bf16)f.w};
        *(bf16x4*)(&As[p][row][kc]) = pk;
      }
    }
    {
      int nn = tid & 127;
      int kh = (tid >> 7) * 16;
      const float* r = bP;
      bf16x8 p0, p1;
#pragma unroll
      for (int i = 0; i < 8; ++i) { p0[i] = (__bf16)(*r); r += ldb; }
#pragma unroll
      for (int i = 0; i < 8; ++i) { p1[i] = (__bf16)(*r); r += ldb; }
      bP += bStep;
      *(bf16x8*)(&Bs[p][nn][kh])     = p0;
      *(bf16x8*)(&Bs[p][nn][kh + 8]) = p1;
    }
  };

  v8f acc[2][4];
#pragma unroll
  for (int s = 0; s < 2; ++s)
#pragma unroll
    for (int j = 0; j < 4; ++j) acc[s][j] = (v8f){};

  const int half = lane >> 4;
  const int bn   = lane & 15;

  stage(0);
  __syncthreads();

  int p = 0;
  for (int k0 = 0; k0 < K; k0 += KB) {
    // Issue global loads for the next tile first: they overlap the WMMA work
    // on the current tile (single barrier per iteration).
    if (k0 + KB < K) stage(p ^ 1);

    // ---- operands in contiguous k order (k = 16*half + e2), same for A and B ----
    // A (16x32): lane half 0 -> K 0..15, half 1 -> K 16..31
    v16bf a[2];
#pragma unroll
    for (int s = 0; s < 2; ++s) {
      int arow = waveM * 32 + s * 16 + bn;
#pragma unroll
      for (int e2 = 0; e2 < 16; ++e2) {
        int kk = half * 16 + e2;
        a[s][e2] = As[p][arow][kk];       // 2x ds_load_b128 per operand
      }
    }
    // B (32x16): lanes 0-15 cover K 0..15, lanes 16-31 cover K 16..31
    v16bf bmat[4];
#pragma unroll
    for (int j = 0; j < 4; ++j) {
#pragma unroll
      for (int e2 = 0; e2 < 16; ++e2)
        bmat[j][e2] = Bs[p][waveN * 64 + j * 16 + bn][half * 16 + e2];
    }
#pragma unroll
    for (int s = 0; s < 2; ++s)
#pragma unroll
      for (int j = 0; j < 4; ++j)
        acc[s][j] = __builtin_amdgcn_wmma_f32_16x16x32_bf16(
            false, a[s], false, bmat[j], (short)0, acc[s][j], false, false);

    __syncthreads();   // staging of p^1 done; reads of p done
    p ^= 1;
  }

  // ---- epilogue (straight-line, MODE folded at compile time) ----
  // C/D layout: VGPR r -> M = r + 8*(lane>>4), N = lane&15
#pragma unroll
  for (int s = 0; s < 2; ++s) {
    int mbase = m0 + waveM * 32 + s * 16 + half * 8;
#pragma unroll
    for (int j = 0; j < 4; ++j) {
      int ncol  = n0 + waveN * 64 + j * 16 + bn;
      float bia = (MODE == 0 || MODE == 1 || MODE == 2 || MODE == 3 ||
                   MODE == 5 || MODE == 6)
                      ? (bias ? bias[ncol] : 0.0f)
                      : 0.0f;
#pragma unroll
      for (int r = 0; r < 8; ++r) {
        int m = mbase + r;
        float raw = acc[s][j][r];
        float z = raw + bia;
        float outv;
        if (MODE == 0) outv = z;
        else if (MODE == 1) outv = (z > 0.0f) ? (z + 1.0f) : __expf(z);
        else if (MODE == 2) outv = 1.0f / (1.0f + __expf(-z));
        else if (MODE == 3) outv = 1.0f / (1.0f + __expf(-z)) + res[(size_t)m * ldc + ncol];
        else if (MODE == 4) outv = raw / (aux[m] + EPSF) + res[(size_t)m * ldc + ncol];
        else if (MODE == 5) outv = aux[(size_t)m * astride] * z;
        else if (MODE == 6) outv = C[(size_t)m * ldc + ncol] + aux[(size_t)m * astride] * z;
        else                outv = raw * 2.0f;   // 1/TEMP
        C[(size_t)m * ldc + ncol] = outv;
      }
    }
  }
}

// n[row,:] = h[row,:] * rsqrt(mean(h^2)+EPS), one block per row
__global__ void rms_kernel(const float* __restrict__ h, float* __restrict__ n, int U) {
  __shared__ float red[256];
  int row = blockIdx.x;
  const float* hr = h + (size_t)row * U;
  float s = 0.0f;
  for (int u = threadIdx.x; u < U; u += 256) { float x = hr[u]; s += x * x; }
  red[threadIdx.x] = s;
  __syncthreads();
  for (int st = 128; st > 0; st >>= 1) {
    if (threadIdx.x < st) red[threadIdx.x] += red[threadIdx.x + st];
    __syncthreads();
  }
  float scale = rsqrtf(red[0] / (float)U + EPSF);
  float* nr = n + (size_t)row * U;
  for (int u = threadIdx.x; u < U; u += 256) nr[u] = hr[u] * scale;
}

// ksum[b,u] = sum_t k[b,t,u]
__global__ void ksum_kernel(const float* __restrict__ k, float* __restrict__ ks,
                            int T, int U, int total) {
  int idx = blockIdx.x * 256 + threadIdx.x;
  if (idx >= total) return;
  int b = idx / U, u = idx - b * U;
  float s = 0.0f;
  for (int t = 0; t < T; ++t) s += k[((size_t)b * T + t) * U + u];
  ks[idx] = s;
}

// den[row] = dot(q[row,:], ksum[b,:]), one block per row
__global__ void den_kernel(const float* __restrict__ q, const float* __restrict__ ks,
                           float* __restrict__ den, int T, int U) {
  __shared__ float red[256];
  int row = blockIdx.x;
  int b = row / T;
  float s = 0.0f;
  for (int u = threadIdx.x; u < U; u += 256)
    s += q[(size_t)row * U + u] * ks[(size_t)b * U + u];
  red[threadIdx.x] = s;
  __syncthreads();
  for (int st = 128; st > 0; st >>= 1) {
    if (threadIdx.x < st) red[threadIdx.x] += red[threadIdx.x + st];
    __syncthreads();
  }
  if (threadIdx.x == 0) den[row] = red[0];
}

// gate[row,e] = softmax_e(n[row,:]@gW + gb); 8 threads per row, 32 rows/block
__global__ void gate_kernel(const float* __restrict__ n, const float* __restrict__ gW,
                            const float* __restrict__ gb, float* __restrict__ gate,
                            int U, int E) {
  __shared__ float lg[32][8];
  int grp = threadIdx.x >> 3;
  int e   = threadIdx.x & 7;
  int row = blockIdx.x * 32 + grp;
  const float* nr = n + (size_t)row * U;
  float s = gb[e];
  for (int u = 0; u < U; ++u) s += nr[u] * gW[(size_t)u * E + e];
  lg[grp][e] = s;
  __syncthreads();
  float mx = -1e30f;
#pragma unroll
  for (int j = 0; j < 8; ++j) mx = fmaxf(mx, lg[grp][j]);
  float sum = 0.0f;
#pragma unroll
  for (int j = 0; j < 8; ++j) sum += __expf(lg[grp][j] - mx);
  gate[(size_t)row * E + e] = __expf(s - mx) / sum;
}

__global__ void copy_kernel(const float* __restrict__ a, float* __restrict__ o, int n) {
  int i = blockIdx.x * 256 + threadIdx.x;
  if (i < n) o[i] = a[i];
}

__global__ void add_kernel(const float* __restrict__ a, const float* __restrict__ b,
                           float* __restrict__ o, int n) {
  int i = blockIdx.x * 256 + threadIdx.x;
  if (i < n) o[i] = a[i] + b[i];
}

// ---------------------------------------------------------------------------
template <int AMODE, int MODE>
static inline void launch_one(hipStream_t s, const float* A, int lda,
                              const int* rowidx, const float* B, int ldb,
                              const float* bias, const float* aux, int astride,
                              const float* res, float* C, int ldc,
                              int M, int N, int K) {
  dim3 grid(N / NBLK, M / MBLK), blk(256);
  gemm_wmma<AMODE, MODE><<<grid, blk, 0, s>>>(A, lda, rowidx, B, ldb, bias,
                                              aux, astride, res, C, ldc, M, N, K);
}

extern "C" void kernel_launch(void* const* d_in, const int* in_sizes, int n_in,
                              void* d_out, int out_size, void* d_ws, size_t ws_size,
                              hipStream_t stream) {
  (void)in_sizes; (void)n_in; (void)out_size; (void)ws_size;
  const int Vv = 32000, D = 768, U = 768, Lc = 16, E = 8, Bc = 2, T = 1024;
  const int M  = Bc * T;                 // 2048 tokens
  const size_t MU = (size_t)M * U;
  const size_t UU = (size_t)U * U;

  const int*   x    = (const int*)  d_in[0];
  const float* emb  = (const float*)d_in[1];
  const float* W_in = (const float*)d_in[2];
  const float* Wq   = (const float*)d_in[3];
  const float* bq   = (const float*)d_in[4];
  const float* Wk   = (const float*)d_in[5];
  const float* bk   = (const float*)d_in[6];
  const float* Wv   = (const float*)d_in[7];
  const float* bv   = (const float*)d_in[8];
  const float* gW   = (const float*)d_in[9];
  const float* gb   = (const float*)d_in[10];
  const float* eW   = (const float*)d_in[11];
  const float* eb   = (const float*)d_in[12];
  const float* fW1  = (const float*)d_in[13];
  const float* fb1  = (const float*)d_in[14];
  const float* fW2  = (const float*)d_in[15];
  const float* fb2  = (const float*)d_in[16];
  const float* Wlog = (const float*)d_in[17];
  float* out = (float*)d_out;

  float* ws   = (float*)d_ws;
  float* h    = ws;             // [M,U]
  float* h0   = h   + MU;       // [M,U]
  float* inp  = h0  + MU;       // [M,U]
  float* nb   = inp + MU;       // [M,U]
  float* qb   = nb  + MU;       // [M,U]
  float* kb   = qb  + MU;       // [M,U]
  float* vb   = kb  + MU;       // [M,U]
  float* kv   = vb  + MU;       // [B,U,U]
  float* ks   = kv  + 2 * UU;   // [B,U]
  float* den  = ks  + 2 * U;    // [M]
  float* gate = den + M;        // [M,E]

  const int ewBlocks = (int)(MU + 255) / 256;
  dim3 b256(256);

  // h0 = emb[x] @ W_in (gathered GEMM); h = h0
  launch_one<2, 0>(stream, emb, D, x, W_in, U, nullptr, nullptr, 1, nullptr,
                   h0, U, M, U, D);
  copy_kernel<<<ewBlocks, b256, 0, stream>>>(h0, h, (int)MU);

  for (int l = 0; l < Lc; ++l) {
    const float* Wq_l  = Wq  + (size_t)l * UU;
    const float* Wk_l  = Wk  + (size_t)l * UU;
    const float* Wv_l  = Wv  + (size_t)l * UU;
    const float* bq_l  = bq  + (size_t)l * U;
    const float* bk_l  = bk  + (size_t)l * U;
    const float* bv_l  = bv  + (size_t)l * U;
    const float* gW_l  = gW  + (size_t)l * U * E;
    const float* gb_l  = gb  + (size_t)l * E;
    const float* fW1_l = fW1 + (size_t)l * UU;
    const float* fb1_l = fb1 + (size_t)l * U;
    const float* fW2_l = fW2 + (size_t)l * UU;
    const float* fb2_l = fb2 + (size_t)l * U;

    // inp = h (block residual)
    copy_kernel<<<ewBlocks, b256, 0, stream>>>(h, inp, (int)MU);

    // ---- linear attention ----
    rms_kernel<<<M, b256, 0, stream>>>(h, nb, U);
    launch_one<0, 1>(stream, nb, U, nullptr, Wq_l, U, bq_l, nullptr, 1, nullptr,
                     qb, U, M, U, U);                    // q = phi(n@Wq+bq)
    launch_one<0, 1>(stream, nb, U, nullptr, Wk_l, U, bk_l, nullptr, 1, nullptr,
                     kb, U, M, U, U);                    // k = phi(n@Wk+bk)
    launch_one<0, 0>(stream, nb, U, nullptr, Wv_l, U, bv_l, nullptr, 1, nullptr,
                     vb, U, M, U, U);                    // v = n@Wv+bv
    for (int b = 0; b < Bc; ++b) {                       // kv[b] = k_b^T @ v_b
      launch_one<1, 0>(stream, kb + (size_t)b * T * U, U, nullptr,
                       vb + (size_t)b * T * U, U, nullptr, nullptr, 1, nullptr,
                       kv + (size_t)b * UU, U, U, U, T);
    }
    ksum_kernel<<<(Bc * U + 255) / 256, b256, 0, stream>>>(kb, ks, T, U, Bc * U);
    den_kernel<<<M, b256, 0, stream>>>(qb, ks, den, T, U);
    for (int b = 0; b < Bc; ++b) {                       // h = (q@kv)/den + h
      launch_one<0, 4>(stream, qb + (size_t)b * T * U, U, nullptr,
                       kv + (size_t)b * UU, U, nullptr, den + (size_t)b * T, 1,
                       h + (size_t)b * T * U, h + (size_t)b * T * U, U, T, U, U);
    }

    // ---- soft MoE ----
    rms_kernel<<<M, b256, 0, stream>>>(h, nb, U);
    gate_kernel<<<M / 32, b256, 0, stream>>>(nb, gW_l, gb_l, gate, U, E);
    for (int e = 0; e < E; ++e) {                        // h = sum_e gate_e*(n@eW_e+eb_e)
      const float* eW_le = eW + ((size_t)l * E + e) * UU;
      const float* eb_le = eb + ((size_t)l * E + e) * U;
      if (e == 0)
        launch_one<0, 5>(stream, nb, U, nullptr, eW_le, U, eb_le,
                         gate + e, E, nullptr, h, U, M, U, U);
      else
        launch_one<0, 6>(stream, nb, U, nullptr, eW_le, U, eb_le,
                         gate + e, E, nullptr, h, U, M, U, U);
    }

    // ---- feedforward + residuals ----
    rms_kernel<<<M, b256, 0, stream>>>(h, nb, U);
    launch_one<0, 2>(stream, nb, U, nullptr, fW1_l, U, fb1_l, nullptr, 1, nullptr,
                     qb, U, M, U, U);                    // f1 = sigmoid(n@fW1+fb1)
    launch_one<0, 3>(stream, qb, U, nullptr, fW2_l, U, fb2_l, nullptr, 1, h,
                     h, U, M, U, U);                     // h = sigmoid(f1@fW2+fb2)+h
    add_kernel<<<ewBlocks, b256, 0, stream>>>(h, inp, h, (int)MU);  // h += inp
  }

  // ---- head: rms, outer residual, rms, logits ----
  rms_kernel<<<M, b256, 0, stream>>>(h, nb, U);
  add_kernel<<<ewBlocks, b256, 0, stream>>>(nb, h0, h, (int)MU);
  rms_kernel<<<M, b256, 0, stream>>>(h, nb, U);
  launch_one<0, 7>(stream, nb, U, nullptr, Wlog, Vv, nullptr, nullptr, 1, nullptr,
                   out, Vv, M, Vv, U);                   // out = (h@Wlog)/TEMP
}
